// SelectiveScan2D_34385508172149
// MI455X (gfx1250) — compile-verified, matrix operand order checked by reading, not run on hardware
//
#include <hip/hip_runtime.h>
#include <hip/hip_bf16.h>

// ---------------------------------------------------------------------------
// Mamba-style SSM block (no sequential scan in reference -> fully parallel):
//   GEMM1 (bf16 WMMA) -> depthwise conv+SiLU -> GEMM2 (bf16 WMMA)
//   -> softplus/exp/SiLU-gate pointwise -> GEMM3 (bf16 WMMA) -> d_out (f32)
//
// GEMM: 256 threads = 8 wave32; block tile 128(M) x 256(N); K-step 32.
// Each wave owns a 64x64 patch: 4 A-frags x 4 B-frags -> 16 WMMAs per K-step
// from 16 ds_load_b128 (1:1 LDS-load : WMMA ratio).
// ---------------------------------------------------------------------------

#define SS_D_MODEL 1024
#define SS_D_STATE 16
#define SS_D_CONV  4
#define SS_INNER   2048
#define SS_B       2
#define SS_L       2048
#define SS_BL      (SS_B * SS_L)   // 4096 rows for every GEMM

typedef __attribute__((ext_vector_type(16))) __bf16 v16bf;
typedef __attribute__((ext_vector_type(8)))  float  v8f;

union SS_Frag { v16bf v; uint4 q[2]; };

// ---------------- prep: f32 -> bf16 cast (for x) ----------------
__global__ __launch_bounds__(256)
void ss_cast_bf16(const float* __restrict__ src, __bf16* __restrict__ dst, long n) {
    long i = (long)blockIdx.x * blockDim.x + threadIdx.x;
    if (i < n) dst[i] = (__bf16)src[i];
}

// ---------------- prep: W (K x N, row major f32) -> Wt (N x K, bf16) --------
__global__ __launch_bounds__(256)
void ss_transpose_bf16(const float* __restrict__ W, __bf16* __restrict__ Wt,
                       int K, int N) {
    long i = (long)blockIdx.x * blockDim.x + threadIdx.x;
    long total = (long)K * N;
    if (i >= total) return;
    long n = i / K;
    long k = i % K;
    Wt[i] = (__bf16)W[k * (long)N + n];
}

// ---------------- bf16 WMMA GEMM: C[M,N] = A[M,K] * Bt[N,K]^T + bias --------
__global__ __launch_bounds__(256)
void ss_gemm_bf16(const __bf16* __restrict__ A, const __bf16* __restrict__ Bt,
                  const float* __restrict__ bias, float* __restrict__ C,
                  int M, int N, int K) {
    __shared__ alignas(16) __bf16 lA[128][32];   //  8 KB
    __shared__ alignas(16) __bf16 lB[256][32];   // 16 KB

    const int tid  = threadIdx.x;
    const int lane = tid & 31;
    const int wave = tid >> 5;
    const int half = (lane >> 4) & 1;   // 0 -> K 0-7/16-23, 1 -> K 8-15/24-31
    const int lrow = lane & 15;
    const int wm   = wave >> 2;         // 0..1 -> M sub-tile of 64
    const int wn   = wave & 3;          // 0..3 -> N sub-tile of 64

    const long rowA0 = (long)blockIdx.y * 128;
    const long rowB0 = (long)blockIdx.x * 256;

    v8f acc[4][4];
#pragma unroll
    for (int i = 0; i < 4; ++i)
#pragma unroll
        for (int j = 0; j < 4; ++j)
            acc[i][j] = (v8f){};

    for (int kt = 0; kt < K; kt += 32) {
        // stage A tile (128x32): 512 16B-chunks, 2 per thread
#pragma unroll
        for (int c = 0; c < 2; ++c) {
            int cid = tid + c * 256;
            int r   = cid >> 2;
            int kc  = (cid & 3) * 8;
            *(uint4*)&lA[r][kc] =
                *(const uint4*)&A[(rowA0 + r) * (long)K + kt + kc];
        }
        // stage Bt tile (256x32): 1024 16B-chunks, 4 per thread
#pragma unroll
        for (int c = 0; c < 4; ++c) {
            int cid = tid + c * 256;
            int r   = cid >> 2;
            int kc  = (cid & 3) * 8;
            *(uint4*)&lB[r][kc] =
                *(const uint4*)&Bt[(rowB0 + r) * (long)K + kt + kc];
        }
        __syncthreads();

        SS_Frag a[4], b[4];
#pragma unroll
        for (int i = 0; i < 4; ++i) {
            int r = wm * 64 + i * 16 + lrow;
            a[i].q[0] = *(const uint4*)&lA[r][half * 8];
            a[i].q[1] = *(const uint4*)&lA[r][16 + half * 8];
        }
#pragma unroll
        for (int j = 0; j < 4; ++j) {
            int r = wn * 64 + j * 16 + lrow;
            b[j].q[0] = *(const uint4*)&lB[r][half * 8];
            b[j].q[1] = *(const uint4*)&lB[r][16 + half * 8];
        }

#pragma unroll
        for (int i = 0; i < 4; ++i)
#pragma unroll
            for (int j = 0; j < 4; ++j)
                acc[i][j] = __builtin_amdgcn_wmma_f32_16x16x32_bf16(
                    false, a[i].v, false, b[j].v, (short)0, acc[i][j],
                    false, false);
        __syncthreads();
    }

    // epilogue: VGPR r -> lanes 0-15 M=r, lanes 16-31 M=r+8; N = lane&15
#pragma unroll
    for (int i = 0; i < 4; ++i) {
#pragma unroll
        for (int j = 0; j < 4; ++j) {
            long gm0 = rowA0 + wm * 64 + i * 16 + half * 8;
            long gn  = rowB0 + wn * 64 + j * 16 + lrow;
            float bv = bias[gn];
#pragma unroll
            for (int r = 0; r < 8; ++r)
                C[(gm0 + r) * (long)N + gn] = acc[i][j][r] + bv;
        }
    }
}

// ---------------- causal depthwise conv (K=4) + SiLU -> bf16 ----------------
__global__ __launch_bounds__(256)
void ss_conv_silu(const float* __restrict__ xz, const float* __restrict__ conv_w,
                  const float* __restrict__ conv_b, __bf16* __restrict__ xin_bf) {
    long idx = (long)blockIdx.x * blockDim.x + threadIdx.x;
    if (idx >= (long)SS_BL * SS_INNER) return;
    int  ch = (int)(idx % SS_INNER);
    long bl = idx / SS_INNER;
    int  l  = (int)(bl % SS_L);
    long b  = bl / SS_L;

    float acc = conv_b[ch];
#pragma unroll
    for (int k = 0; k < SS_D_CONV; ++k) {
        int ls = l - (SS_D_CONV - 1) + k;
        if (ls >= 0)
            acc += conv_w[ch * SS_D_CONV + k] *
                   xz[((b * SS_L + ls) * (long)(2 * SS_INNER)) + ch];
    }
    float s = acc / (1.f + __expf(-acc));      // SiLU
    xin_bf[idx] = (__bf16)s;
}

// ---------------- softplus(dt), sum_n exp(dt*A), +D, *silu(z) -> bf16 -------
__global__ __launch_bounds__(256)
void ss_pointwise(const __bf16* __restrict__ xin_bf, const float* __restrict__ dtlin,
                  const float* __restrict__ xz, const float* __restrict__ A_log,
                  const float* __restrict__ Dvec, __bf16* __restrict__ y_bf) {
    long idx = (long)blockIdx.x * blockDim.x + threadIdx.x;
    if (idx >= (long)SS_BL * SS_INNER) return;
    int  ch = (int)(idx % SS_INNER);
    long bl = idx / SS_INNER;

    float xv = (float)xin_bf[idx];
    float t  = dtlin[idx];
    float dt = (t > 20.f) ? t : log1pf(__expf(t));  // softplus

    float s = 0.f;
#pragma unroll
    for (int n = 0; n < SS_D_STATE; ++n) {
        float Aval = -__expf(A_log[ch * SS_D_STATE + n]);
        s += __expf(dt * Aval);
    }
    float y = xv * s + xv * Dvec[ch];
    float z = xz[bl * (long)(2 * SS_INNER) + SS_INNER + ch];
    y *= z / (1.f + __expf(-z));                // * silu(z)
    y_bf[idx] = (__bf16)y;
}

// ---------------------------------------------------------------------------
extern "C" void kernel_launch(void* const* d_in, const int* in_sizes, int n_in,
                              void* d_out, int out_size, void* d_ws, size_t ws_size,
                              hipStream_t stream) {
    const float* x      = (const float*)d_in[0];
    const float* W_in   = (const float*)d_in[1];
    const float* b_in   = (const float*)d_in[2];
    const float* conv_w = (const float*)d_in[3];
    const float* conv_b = (const float*)d_in[4];
    const float* A_log  = (const float*)d_in[5];
    const float* Dvec   = (const float*)d_in[6];
    const float* W_dt   = (const float*)d_in[7];
    const float* b_dt   = (const float*)d_in[8];
    const float* W_out  = (const float*)d_in[9];
    const float* b_out  = (const float*)d_in[10];
    float* out = (float*)d_out;

    // workspace carve-up (~164 MB total)
    char* ws = (char*)d_ws;
    size_t off = 0;
    auto carve = [&](size_t bytes) {
        size_t r = off;
        off = (off + bytes + 255) & ~(size_t)255;
        return r;
    };
    __bf16* xbf   = (__bf16*)(ws + carve((size_t)SS_BL * SS_D_MODEL * 2));
    __bf16* WinT  = (__bf16*)(ws + carve((size_t)(2 * SS_INNER) * SS_D_MODEL * 2));
    __bf16* WdtT  = (__bf16*)(ws + carve((size_t)SS_INNER * SS_INNER * 2));
    __bf16* WoutT = (__bf16*)(ws + carve((size_t)SS_D_MODEL * SS_INNER * 2));
    float*  xz    = (float*)(ws + carve((size_t)SS_BL * (2 * SS_INNER) * 4));
    __bf16* xinbf = (__bf16*)(ws + carve((size_t)SS_BL * SS_INNER * 2));
    float*  dtlin = (float*)(ws + carve((size_t)SS_BL * SS_INNER * 4));
    __bf16* ybf   = (__bf16*)(ws + carve((size_t)SS_BL * SS_INNER * 2));
    (void)ws_size;

    // --- prep: casts / weight transposes ---
    {
        long n = (long)SS_BL * SS_D_MODEL;
        ss_cast_bf16<<<(int)((n + 255) / 256), 256, 0, stream>>>(x, xbf, n);
    }
    {
        long n = (long)SS_D_MODEL * (2 * SS_INNER);
        ss_transpose_bf16<<<(int)((n + 255) / 256), 256, 0, stream>>>(
            W_in, WinT, SS_D_MODEL, 2 * SS_INNER);
    }
    {
        long n = (long)SS_INNER * SS_INNER;
        ss_transpose_bf16<<<(int)((n + 255) / 256), 256, 0, stream>>>(
            W_dt, WdtT, SS_INNER, SS_INNER);
    }
    {
        long n = (long)SS_INNER * SS_D_MODEL;
        ss_transpose_bf16<<<(int)((n + 255) / 256), 256, 0, stream>>>(
            W_out, WoutT, SS_INNER, SS_D_MODEL);
    }

    // --- GEMM1: xz = x @ W_in + b_in  (4096 x 4096 x K=1024) ---
    {
        dim3 grid((2 * SS_INNER) / 256, SS_BL / 128);
        ss_gemm_bf16<<<grid, 256, 0, stream>>>(xbf, WinT, b_in, xz,
                                               SS_BL, 2 * SS_INNER, SS_D_MODEL);
    }

    // --- conv + SiLU ---
    {
        long n = (long)SS_BL * SS_INNER;
        ss_conv_silu<<<(int)((n + 255) / 256), 256, 0, stream>>>(
            xz, conv_w, conv_b, xinbf);
    }

    // --- GEMM2: dt_lin = x_in @ W_dt + b_dt  (4096 x 2048 x K=2048) ---
    {
        dim3 grid(SS_INNER / 256, SS_BL / 128);
        ss_gemm_bf16<<<grid, 256, 0, stream>>>(xinbf, WdtT, b_dt, dtlin,
                                               SS_BL, SS_INNER, SS_INNER);
    }

    // --- pointwise SSM epilogue ---
    {
        long n = (long)SS_BL * SS_INNER;
        ss_pointwise<<<(int)((n + 255) / 256), 256, 0, stream>>>(
            xinbf, dtlin, xz, A_log, Dvec, ybf);
    }

    // --- GEMM3: out = y @ W_out + b_out  (4096 x 1024 x K=2048) ---
    {
        dim3 grid(SS_D_MODEL / 256, SS_BL / 128);
        ss_gemm_bf16<<<grid, 256, 0, stream>>>(ybf, WoutT, b_out, out,
                                               SS_BL, SS_D_MODEL, SS_INNER);
    }
}
